// AnchorStripeAttention_9277129359821
// MI455X (gfx1250) — matrix operand order repaired and checked by reading, not compiled, not run on hardware
//
#include <hip/hip_runtime.h>

typedef _Float16 h16 __attribute__((ext_vector_type(16)));
typedef _Float16 h8  __attribute__((ext_vector_type(8)));
typedef float    f8  __attribute__((ext_vector_type(8)));

#define T_FR   5
#define NHEADS 4
#define HD     32
#define N1TOK  640      // 5*8*16 window tokens
#define N2TOK  160      // 5*4*8 anchor tokens
#define LDR    40       // padded f16 row stride (80B, 16B-aligned rows)
#define NWAVE  10
#define NTAB   2277     // 9*11*23 bias table rows
#define LOG2E  1.4426950408889634f
// LDS map (halfs): sKB 20480 | sV 25600 | sA 6400 | sAB 5120 | sY 6400 | sP 5120 | +640 floats
#define OFF_KB 0
#define OFF_V  20480
#define OFF_A  46080
#define OFF_AB 52480
#define OFF_Y  57600
#define OFF_P  64000
#define SMEM_BYTES (69120 * 2 + 640 * 4)   // 140800

// ---------------- bias MLP: table(3) -> relu(512) -> 4 heads ----------------
__global__ void bias_mlp_kernel(const float* __restrict__ w1, const float* __restrict__ b1,
                                const float* __restrict__ w2, float* __restrict__ biasTab) {
    int idx = blockIdx.x * blockDim.x + threadIdx.x;
    if (idx >= NTAB) return;
    int ww = idx % 23; int rest = idx / 23;
    int hh = rest % 11; int ti = rest / 11;
    float c[3];
    c[0] = (float)(ti - 4) * (1.0f / 4.0f);
    c[1] = (float)(hh - 5) * (1.0f / 5.0f);
    c[2] = (float)(ww - 11) * (1.0f / 11.0f);
    const float inv_log2_8 = 1.0f / 3.0f;
#pragma unroll
    for (int d = 0; d < 3; ++d) {
        float x = c[d] * 8.0f;
        c[d] = copysignf(log2f(fabsf(x) + 1.0f) * inv_log2_8, x);
    }
    float acc0 = 0.f, acc1 = 0.f, acc2 = 0.f, acc3 = 0.f;
    for (int kk = 0; kk < 512; ++kk) {
        float hsum = fmaf(c[0], w1[kk], fmaf(c[1], w1[512 + kk], fmaf(c[2], w1[1024 + kk], b1[kk])));
        hsum = fmaxf(hsum, 0.0f);
        acc0 = fmaf(hsum, w2[kk * 4 + 0], acc0);
        acc1 = fmaf(hsum, w2[kk * 4 + 1], acc1);
        acc2 = fmaf(hsum, w2[kk * 4 + 2], acc2);
        acc3 = fmaf(hsum, w2[kk * 4 + 3], acc3);
    }
    biasTab[idx * 4 + 0] = acc0;
    biasTab[idx * 4 + 1] = acc1;
    biasTab[idx * 4 + 2] = acc2;
    biasTab[idx * 4 + 3] = acc3;
}

// ---------------- WMMA + fragment helpers (wave32, 16x16x32 f16) ----------------
__device__ __forceinline__ f8 fzero() {
    f8 z;
#pragma unroll
    for (int i = 0; i < 8; ++i) z[i] = 0.0f;
    return z;
}

__device__ __forceinline__ f8 wmma_f16(h16 a, h16 b, f8 c) {
    return __builtin_amdgcn_wmma_f32_16x16x32_f16(false, a, false, b, (short)0, c, false, false);
}

__device__ __forceinline__ h16 cat8(h8 lo, h8 hi2) {
    return __builtin_shufflevector(lo, hi2, 0, 1, 2, 3, 4, 5, 6, 7, 8, 9, 10, 11, 12, 13, 14, 15);
}

// A fragment 16x32 from a row-major tile: two 16B contiguous chunks per lane.
__device__ __forceinline__ h16 loadA(const _Float16* p, int ld, int lrow, int hi) {
    const _Float16* r = p + lrow * ld + 8 * hi;
    return cat8(*(const h8*)r, *(const h8*)(r + 16));
}

// B fragment 32x16 from a tile [K][N] (stride 16): lane = N column, elements over K (table order).
__device__ __forceinline__ h16 loadB_sw(const _Float16* p, int lane) {
    h16 v; int n = lane & 15, h = lane >> 4;
#pragma unroll
    for (int s = 0; s < 8; ++s) { v[s] = p[(8 * h + s) * 16 + n]; v[8 + s] = p[(16 + 8 * h + s) * 16 + n]; }
    return v;
}

// B fragment 32x16 from a row-major tile (lane = N column, elements over K in table order).
__device__ __forceinline__ h16 loadB_row(const _Float16* p, int ld, int lane) {
    h16 v; int n = lane & 15, h = lane >> 4;
#pragma unroll
    for (int s = 0; s < 8; ++s) { v[s] = p[(8 * h + s) * ld + n]; v[8 + s] = p[(16 + 8 * h + s) * ld + n]; }
    return v;
}

// ---- DPP16 butterfly reductions over 16-lane groups, DPP fused into the VALU op ----
// (single instruction per step: v_max_num_f32/v_add_f32 with src0 DPP16 modifier)
__device__ __forceinline__ float rmax16(float v) {
    float t;
    asm("v_max_num_f32 %0, %1, %1 quad_perm:[1,0,3,2] row_mask:0xf bank_mask:0xf bound_ctrl:1"
        : "=v"(t) : "v"(v));
    asm("v_max_num_f32 %0, %1, %1 quad_perm:[2,3,0,1] row_mask:0xf bank_mask:0xf bound_ctrl:1"
        : "=v"(v) : "v"(t));
    asm("v_max_num_f32 %0, %1, %1 row_half_mirror row_mask:0xf bank_mask:0xf bound_ctrl:1"
        : "=v"(t) : "v"(v));
    asm("v_max_num_f32 %0, %1, %1 row_mirror row_mask:0xf bank_mask:0xf bound_ctrl:1"
        : "=v"(v) : "v"(t));
    return v;
}
__device__ __forceinline__ float rsum16(float v) {
    float t;
    asm("v_add_f32 %0, %1, %1 quad_perm:[1,0,3,2] row_mask:0xf bank_mask:0xf bound_ctrl:1"
        : "=v"(t) : "v"(v));
    asm("v_add_f32 %0, %1, %1 quad_perm:[2,3,0,1] row_mask:0xf bank_mask:0xf bound_ctrl:1"
        : "=v"(v) : "v"(t));
    asm("v_add_f32 %0, %1, %1 row_half_mirror row_mask:0xf bank_mask:0xf bound_ctrl:1"
        : "=v"(t) : "v"(v));
    asm("v_add_f32 %0, %1, %1 row_mirror row_mask:0xf bank_mask:0xf bound_ctrl:1"
        : "=v"(v) : "v"(t));
    return v;
}

// ---------------- main attention kernel: 1 block = (window, head) ----------------
__global__ __launch_bounds__(320)
void anchor_stripe_attn_kernel(const float* __restrict__ qkv,
                               const float* __restrict__ anchor,
                               const float* __restrict__ mask_a2w,
                               const float* __restrict__ mask_w2a,
                               const float* __restrict__ biasTab,
                               float* __restrict__ out) {
    extern __shared__ _Float16 smem[];
    _Float16* sKB = smem + OFF_KB;   // 40 tiles x [32 lanes(K)][16 (N)] l2n(k) swizzled
    _Float16* sV  = smem + OFF_V;    // 640 x 40 row-major v
    _Float16* sA  = smem + OFF_A;    // 160 x 40 row-major l2n(anchor) (A-frags)
    _Float16* sAB = smem + OFF_AB;   // 10 tiles x [32][16] l2n(anchor)^T swizzled (B-frags)
    _Float16* sY  = smem + OFF_Y;    // 160 x 40 stage-1 output
    _Float16* sP  = smem + OFF_P;    // 10 waves x 16x32 C->A transpose scratch
    float* qInvN  = (float*)(smem + OFF_P + NWAVE * 512);  // 640 q row inv norms

    const int win  = blockIdx.x >> 2;
    const int head = blockIdx.x & 3;
    const int hBlk = win >> 3;
    const int wBlk = win & 7;
    const int tid  = threadIdx.x;

    // -------- preprocessing: gather + l2-normalize into LDS --------
    for (int n2 = tid; n2 < N2TOK; n2 += 320) {
        int t = n2 >> 5, i2 = (n2 >> 3) & 3, j2 = n2 & 7;
        int ah = (hBlk * 4 + i2 + 2) & 63;
        int aw = (wBlk * 8 + j2 + 4) & 63;
        const float* src = anchor + ((size_t)((t * 64 + ah) * 64 + aw)) * 128 + head * HD;
        float v[HD]; float ss = 0.f;
#pragma unroll
        for (int d = 0; d < HD; ++d) { v[d] = src[d]; ss += v[d] * v[d]; }
        float inv = rsqrtf(fmaxf(ss, 1e-24f));
        _Float16* rowp = sA + n2 * LDR;
#pragma unroll
        for (int c = 0; c < 4; ++c) {
            h8 o;
#pragma unroll
            for (int d = 0; d < 8; ++d) o[d] = (_Float16)(v[c * 8 + d] * inv);
            *(h8*)(rowp + c * 8) = o;
        }
        _Float16* swz = sAB + (n2 >> 4) * 512 + (n2 & 15);
#pragma unroll
        for (int d = 0; d < HD; ++d) swz[d * 16] = (_Float16)(v[d] * inv);
    }
    for (int n1 = tid; n1 < N1TOK; n1 += 320) {
        int t = n1 >> 7, i = (n1 >> 4) & 7, j = n1 & 15;
        int hg = (hBlk * 8 + i + 4) & 127;
        int wg = (wBlk * 16 + j + 8) & 127;
        const float* row = qkv + ((size_t)(t * 16384 + hg * 128 + wg)) * 384 + head * HD;
        float kv[HD]; float ss = 0.f;
#pragma unroll
        for (int d = 0; d < HD; ++d) { kv[d] = row[128 + d]; ss += kv[d] * kv[d]; }
        float inv = rsqrtf(fmaxf(ss, 1e-24f));
        _Float16* swz = sKB + (n1 >> 4) * 512 + (n1 & 15);
#pragma unroll
        for (int d = 0; d < HD; ++d) swz[d * 16] = (_Float16)(kv[d] * inv);
        float qs = 0.f;
        _Float16* vrow = sV + n1 * LDR;
#pragma unroll
        for (int c = 0; c < 4; ++c) {
            h8 o;
#pragma unroll
            for (int d = 0; d < 8; ++d) {
                float x = row[c * 8 + d]; qs += x * x;
                o[d] = (_Float16)row[256 + c * 8 + d];
            }
            *(h8*)(vrow + c * 8) = o;
        }
        qInvN[n1] = rsqrtf(fmaxf(qs, 1e-24f));
    }
    __syncthreads();

    const int wave = tid >> 5;
    const int lane = tid & 31;
    const int lrow = lane & 15;
    const int hi   = lane >> 4;
    _Float16* sPw = sP + wave * 512;

    // ======== stage 1: Y = softmax(a . k^T + bias + mask) . v  (flash, log2 domain) ========
    {
        const int R0 = wave * 16;
        h16 aF = loadA(sA + R0 * LDR, LDR, lrow, hi);
        f8 Y0 = fzero(), Y1 = fzero();
        float mx[8], sm[8];
#pragma unroll
        for (int r = 0; r < 8; ++r) { mx[r] = -3.0e38f; sm[r] = 0.f; }

        for (int c0 = 0; c0 < N1TOK; c0 += 32) {
            h16 b0 = loadB_sw(sKB + (c0 >> 4) * 512, lane);
            h16 b1 = loadB_sw(sKB + ((c0 >> 4) + 1) * 512, lane);
            f8 S0 = wmma_f16(aF, b0, fzero());
            f8 S1 = wmma_f16(aF, b1, fzero());

            int n1a = c0 + lrow, n1b = n1a + 16;
            int t1a = n1a >> 7, i1a = (n1a >> 4) & 7, j1a = n1a & 15;
            int t1b = n1b >> 7, i1b = (n1b >> 4) & 7, j1b = n1b & 15;
#pragma unroll
            for (int r = 0; r < 8; ++r) {
                int n2 = R0 + r + 8 * hi;
                int t2 = n2 >> 5, i2 = (n2 >> 3) & 3, j2 = n2 & 7;
                int ia = ((t2 - t1a + 4) * 11 + (i2 - i1a + 7)) * 23 + (j2 - j1a + 15);
                int ib = ((t2 - t1b + 4) * 11 + (i2 - i1b + 7)) * 23 + (j2 - j1b + 15);
                // log2-domain scores: one mul here saves a mul in every exp below
                S0[r] = (S0[r] + biasTab[ia * 4 + head]
                         + mask_a2w[((size_t)win * N2TOK + n2) * N1TOK + n1a]) * LOG2E;
                S1[r] = (S1[r] + biasTab[ib * 4 + head]
                         + mask_a2w[((size_t)win * N2TOK + n2) * N1TOK + n1b]) * LOG2E;
            }
#pragma unroll
            for (int r = 0; r < 8; ++r) {
                float mnew = rmax16(fmaxf(S0[r], S1[r]));
                float mall = fmaxf(mx[r], mnew);
                float corr = __builtin_amdgcn_exp2f(mx[r] - mall);
                float p0 = __builtin_amdgcn_exp2f(S0[r] - mall);
                float p1 = __builtin_amdgcn_exp2f(S1[r] - mall);
                sm[r] = sm[r] * corr + rsum16(p0 + p1);
                mx[r] = mall;
                Y0[r] *= corr; Y1[r] *= corr;
                sPw[(r + 8 * hi) * 32 + lrow]      = (_Float16)p0;
                sPw[(r + 8 * hi) * 32 + 16 + lrow] = (_Float16)p1;
            }
            h16 pF = loadA(sPw, 32, lrow, hi);
            h16 v0 = loadB_row(sV + c0 * LDR, LDR, lane);
            h16 v1 = loadB_row(sV + c0 * LDR + 16, LDR, lane);
            Y0 = wmma_f16(pF, v0, Y0);
            Y1 = wmma_f16(pF, v1, Y1);
        }
#pragma unroll
        for (int r = 0; r < 8; ++r) {
            float inv = 1.0f / sm[r];
            int n2 = R0 + r + 8 * hi;
            sY[n2 * LDR + lrow]      = (_Float16)(Y0[r] * inv);
            sY[n2 * LDR + 16 + lrow] = (_Float16)(Y1[r] * inv);
        }
    }
    __syncthreads();

    // ======== stage 2: out = softmax(q . a^T + bias + mask) . Y ========
    for (int mt = wave * 4; mt < wave * 4 + 4; ++mt) {
        const int R = mt * 16;
        int n1q = R + lrow;
        int tq = n1q >> 7, iq = (n1q >> 4) & 7, jq = n1q & 15;
        int hgq = (hBlk * 8 + iq + 4) & 127;
        int wgq = (wBlk * 16 + jq + 8) & 127;
        const float* qrow = qkv + ((size_t)(tq * 16384 + hgq * 128 + wgq)) * 384 + head * HD;
        float qin = qInvN[n1q];
        h16 qF;
#pragma unroll
        for (int i = 0; i < 4; ++i) {
            int k = 2 * i + 8 * hi;
            qF[2 * i]     = (_Float16)(qrow[k] * qin);
            qF[2 * i + 1] = (_Float16)(qrow[k + 1] * qin);
            qF[8 + 2 * i] = (_Float16)(qrow[16 + k] * qin);
            qF[9 + 2 * i] = (_Float16)(qrow[16 + k + 1] * qin);
        }
        f8 S[10];
#pragma unroll
        for (int jt = 0; jt < 10; ++jt) {
            h16 b = loadB_sw(sAB + jt * 512, lane);
            S[jt] = wmma_f16(qF, b, fzero());
        }
#pragma unroll
        for (int r = 0; r < 8; ++r) {
            int n1r = R + r + 8 * hi;
            int t1 = n1r >> 7, i1 = (n1r >> 4) & 7, j1 = n1r & 15;
#pragma unroll
            for (int jt = 0; jt < 10; ++jt) {
                int n2 = jt * 16 + lrow;
                int t2 = n2 >> 5, i2 = (n2 >> 3) & 3, j2 = n2 & 7;
                int idx = ((t1 - t2 + 4) * 11 + (i1 - i2 + 3)) * 23 + (j1 - j2 + 7);
                S[jt][r] = (S[jt][r] + biasTab[idx * 4 + head]
                            + mask_w2a[((size_t)win * N1TOK + n1r) * N2TOK + n2]) * LOG2E;
            }
        }
        float inv_[8];
#pragma unroll
        for (int r = 0; r < 8; ++r) {
            float m = S[0][r];
#pragma unroll
            for (int jt = 1; jt < 10; ++jt) m = fmaxf(m, S[jt][r]);
            m = rmax16(m);
            float s = 0.f;
#pragma unroll
            for (int jt = 0; jt < 10; ++jt) {
                float e = __builtin_amdgcn_exp2f(S[jt][r] - m);
                S[jt][r] = e; s += e;
            }
            inv_[r] = 1.0f / rsum16(s);
        }
        f8 O0 = fzero(), O1 = fzero();
#pragma unroll
        for (int kc = 0; kc < 5; ++kc) {
#pragma unroll
            for (int r = 0; r < 8; ++r) {
                sPw[(r + 8 * hi) * 32 + lrow]      = (_Float16)(S[2 * kc][r] * inv_[r]);
                sPw[(r + 8 * hi) * 32 + 16 + lrow] = (_Float16)(S[2 * kc + 1][r] * inv_[r]);
            }
            h16 pF = loadA(sPw, 32, lrow, hi);
            h16 y0 = loadB_row(sY + kc * 32 * LDR, LDR, lane);
            h16 y1 = loadB_row(sY + kc * 32 * LDR + 16, LDR, lane);
            O0 = wmma_f16(pF, y0, O0);
            O1 = wmma_f16(pF, y1, O1);
        }
#pragma unroll
        for (int r = 0; r < 8; ++r) {
            int n1r = R + r + 8 * hi;
            int t = n1r >> 7, i = (n1r >> 4) & 7, j = n1r & 15;
            int hg = (hBlk * 8 + i + 4) & 127;
            int wg = (wBlk * 16 + j + 8) & 127;
            float* dst = out + ((size_t)(t * 16384 + hg * 128 + wg)) * 128 + head * HD;
            dst[lrow]      = O0[r];
            dst[16 + lrow] = O1[r];
        }
    }
}

extern "C" void kernel_launch(void* const* d_in, const int* in_sizes, int n_in,
                              void* d_out, int out_size, void* d_ws, size_t ws_size,
                              hipStream_t stream) {
    (void)in_sizes; (void)n_in; (void)out_size; (void)ws_size;
    const float* qkv      = (const float*)d_in[0];
    const float* anchor   = (const float*)d_in[1];
    const float* mask_a2w = (const float*)d_in[2];
    const float* mask_w2a = (const float*)d_in[3];
    const float* w1       = (const float*)d_in[4];
    const float* b1       = (const float*)d_in[5];
    const float* w2       = (const float*)d_in[6];
    float* out     = (float*)d_out;
    float* biasTab = (float*)d_ws;   // 2277*4 floats

    bias_mlp_kernel<<<dim3(9), dim3(256), 0, stream>>>(w1, b1, w2, biasTab);
    anchor_stripe_attn_kernel<<<dim3(512), dim3(320), SMEM_BYTES, stream>>>(
        qkv, anchor, mask_a2w, mask_w2a, biasTab, out);
}
